// CausalSelfAttention_62895501083199
// MI455X (gfx1250) — compile-verified, matrix operand order checked by reading, not run on hardware
//
#include <hip/hip_runtime.h>
#include <hip/hip_bf16.h>
#include <math.h>

// ---------------------------------------------------------------------------
// CDNA5 / gfx1250 causal self-attention (B=2,T=2048,C=1024,H=16,D=64).
// Compute-bound (~70 GFLOP vs ~100MB of traffic => ~5us HBM time at 23.3TB/s),
// so all matmuls go through v_wmma_f32_16x16x32_f16 (f16 in, f32 accumulate).
// Weight (B-operand) tiles are staged into LDS with the Tensor Data Mover
// (tensor_load_to_lds + s_wait_tensorcnt), shared by the 4 waves of a block
// -> 4x less B traffic and the CDNA5 async-tensor path is exercised.
// ---------------------------------------------------------------------------

typedef __attribute__((ext_vector_type(16))) _Float16 v16h;
typedef __attribute__((ext_vector_type(8)))  _Float16 h8;
typedef __attribute__((ext_vector_type(8)))  float    v8f;
typedef __attribute__((ext_vector_type(4)))  unsigned int v4u;
typedef __attribute__((ext_vector_type(8)))  int v8i;
typedef __attribute__((ext_vector_type(4)))  int v4i;

constexpr int Bc = 2;
constexpr int Tc = 2048;
constexpr int Cc = 1024;
constexpr int Hc = 16;
constexpr int Dc = 64;
constexpr int Mc = Bc * Tc;   // 4096 rows in the flattened GEMMs

__device__ __forceinline__ v16h frag_cat(h8 lo, h8 hi) {
  return __builtin_shufflevector(lo, hi, 0,1,2,3,4,5,6,7,8,9,10,11,12,13,14,15);
}

// Load one 16-bit A/B WMMA fragment (K=32 slice) for this lane.
// 16-bit layout: lanes 0-15 hold K groups {0..7, 16..23}; lanes 16-31 hold
// {8..15, 24..31} -> per lane two contiguous 16-byte loads.
__device__ __forceinline__ v16h load_frag(const _Float16* rowptr, int halfsel) {
  h8 lo = *(const h8*)(rowptr + halfsel * 8);
  h8 hi = *(const h8*)(rowptr + 16 + halfsel * 8);
  return frag_cat(lo, hi);
}

__device__ __forceinline__ v8f wmma_f16(v16h a, v16h b, v8f c) {
  return __builtin_amdgcn_wmma_f32_16x16x32_f16(false, a, false, b,
                                                (short)0, c, false, false);
}

__device__ __forceinline__ float red_max16(float v) {
#pragma unroll
  for (int m = 1; m < 16; m <<= 1) v = fmaxf(v, __shfl_xor(v, m, 32));
  return v;
}
__device__ __forceinline__ float red_sum16(float v) {
#pragma unroll
  for (int m = 1; m < 16; m <<= 1) v += __shfl_xor(v, m, 32);
  return v;
}

// TDM: async-load a 2D f16 tile (tileK x tileN rows) from global (row stride
// strideK halfs) into LDS at ldsAddr, packed row-major tileN x tileK halfs.
// D# layout per CDNA5 ISA (2-D tensor: groups
// 2/3 zero).  clang-23 / therock-10.0 6-arg builtin form:
// (v4u g0, v8i g1, v4i g2, v4i g3, v8i, i32 cpol).
__device__ __forceinline__ void tdm_load_tile_f16(const _Float16* gsrc,
                                                  unsigned ldsAddr,
                                                  int tileK, int tileN,
                                                  int dimK, int dimN,
                                                  int strideK) {
  unsigned long long ga = (unsigned long long)(uintptr_t)gsrc;
  v4u g0;
  g0[0] = 1u;                                   // count=1, user mode, no gather
  g0[1] = ldsAddr;                              // lds_addr (bytes)
  g0[2] = (unsigned)(ga & 0xFFFFFFFFull);       // global_addr[31:0]
  g0[3] = (unsigned)((ga >> 32) & 0x1FFFFFFull) // global_addr[56:32]
          | (2u << 30);                         // type=2 ("image")
  v8i g1;
  g1[0] = (int)(1u << 16);                      // data_size=1 -> 2 bytes
  g1[1] = (int)((unsigned)(dimK & 0xFFFF) << 16);        // tensor_dim0[15:0]
  g1[2] = (int)(((unsigned)(dimN & 0xFFFF) << 16) |      // tensor_dim1[15:0]
                ((unsigned)dimK >> 16));                 // tensor_dim0[31:16]
  g1[3] = (int)(((unsigned)(tileK & 0xFFFF) << 16) |     // tile_dim0
                ((unsigned)dimN >> 16));                 // tensor_dim1[31:16]
  g1[4] = (int)(unsigned)(tileN & 0xFFFF);               // tile_dim1 (tile_dim2=0)
  g1[5] = strideK;                                       // tensor_dim0_stride[31:0]
  g1[6] = 0;                                             // stride0[47:32], stride1[15:0]
  g1[7] = 0;                                             // stride1[47:16]
  v4i gz4 = {0, 0, 0, 0};
  v8i gz8 = {0, 0, 0, 0, 0, 0, 0, 0};
  __builtin_amdgcn_tensor_load_to_lds(g0, g1, gz4, gz4, gz8, 0);
}

// ---------------------------------------------------------------------------
// fp32 -> f16 convert
// ---------------------------------------------------------------------------
__global__ void cvt_f16_kernel(const float* __restrict__ in,
                               _Float16* __restrict__ out, int n) {
  int i = blockIdx.x * blockDim.x + threadIdx.x;
  if (i < n) out[i] = (_Float16)in[i];
}

// W (K x N row-major, fp32) -> Wt (N x K row-major, f16) so B-fragments read
// contiguous K-runs.
__global__ void transpose_cvt_kernel(const float* __restrict__ W,
                                     _Float16* __restrict__ Wt, int K, int N) {
  int i = blockIdx.x * blockDim.x + threadIdx.x;
  if (i < N * K) {
    int n = i / K, k = i % K;
    Wt[i] = (_Float16)W[(size_t)k * N + n];
  }
}

// ---------------------------------------------------------------------------
// Block-tile GEMM: 4 waves/block, block tile 64(M) x 64(N); each wave owns
// 16 x 64 (4 f32 accumulators).  B tile (64 N-rows x 32 K) staged into LDS by
// the TDM once per k-step and shared by all 4 waves.
// mode 0: v    -> store f16 transposed (B,H,D,T)
// mode 1: q/k  -> fuse RoPE + RMSNorm, store f16 (B,H,T,D)
// mode 2: proj -> store fp32 row-major (M,N) to d_out
// ---------------------------------------------------------------------------
__global__ void gemm16_kernel(const _Float16* __restrict__ A,
                              const _Float16* __restrict__ Bt,
                              _Float16* __restrict__ out16,
                              float* __restrict__ outf,
                              const float* __restrict__ cosp,
                              const float* __restrict__ sinp,
                              int M, int N, int K, int mode) {
  __shared__ _Float16 btile[64 * 32];   // 4 KB staged B tile
  const int lane    = threadIdx.x & 31;
  const int halfsel = lane >> 4;        // which 16-lane half
  const int l15     = lane & 15;
  const int wv      = threadIdx.x >> 5;
  const int ntn     = N >> 6;           // N-tiles of 64
  const int mb = blockIdx.x / ntn, nt = blockIdx.x % ntn;
  const int m0 = mb * 64 + 16 * wv;     // per-wave 16-row slice of block tile
  const int n0 = nt * 64;

  v8f acc[4];
  v8f zero = {};
#pragma unroll
  for (int j = 0; j < 4; ++j) acc[j] = zero;

  const _Float16* arow = A + (size_t)(m0 + l15) * K;
  const unsigned ldsB = (unsigned)(uintptr_t)(void*)btile;

  for (int k0 = 0; k0 < K; k0 += 32) {
    if (wv == 0) {
      // One DMA per block: Bt[n0 .. n0+63][k0 .. k0+31] -> btile[64][32]
      tdm_load_tile_f16(Bt + (size_t)n0 * K + k0, ldsB,
                        /*tileK=*/32, /*tileN=*/64, K, N, K);
      __builtin_amdgcn_s_wait_tensorcnt(0);
    }
    __syncthreads();                     // tile visible to all waves

    __builtin_prefetch(arow + k0 + 256, 0, 0);   // global_prefetch_b8
    v16h af = load_frag(arow + k0, halfsel);
#pragma unroll
    for (int j = 0; j < 4; ++j) {
      // B fragment from LDS: row (16j + l15), 32-half row stride.
      v16h bf = load_frag(btile + (size_t)(16 * j + l15) * 32, halfsel);
      acc[j] = wmma_f16(af, bf, acc[j]);
    }
    __syncthreads();                     // protect btile before next DMA
  }

  // C/D layout: element r of lane -> row m0 + r + 8*halfsel, col n0+16j+l15.
  if (mode == 1) {
    const int h = n0 >> 6;
#pragma unroll
    for (int r = 0; r < 8; ++r) {
      const int m = m0 + r + 8 * halfsel;
      const int b = m / Tc, t = m % Tc;
      // RoPE: pair (d, d+32) == accumulators (j, j+2), same lane/element.
#pragma unroll
      for (int j = 0; j < 2; ++j) {
        const int dp = 16 * j + l15;            // 0..31
        const float cv = cosp[t * Dc + dp];
        const float sv = sinp[t * Dc + dp];
        const float x1 = acc[j][r], x2 = acc[j + 2][r];
        acc[j][r]     = x1 * cv - x2 * sv;
        acc[j + 2][r] = x1 * sv + x2 * cv;
      }
      // RMSNorm over the 64-wide head row (4 accs x 16 lanes of this half).
      float ss = 0.f;
#pragma unroll
      for (int j = 0; j < 4; ++j) ss += acc[j][r] * acc[j][r];
      ss = red_sum16(ss);
      const float sc = rsqrtf(ss * (1.0f / 64.0f) + 1e-6f);
#pragma unroll
      for (int j = 0; j < 4; ++j) {
        const float y = acc[j][r] * sc;
        out16[(((size_t)b * Hc + h) * Tc + t) * Dc + 16 * j + l15] = (_Float16)y;
      }
    }
  } else if (mode == 0) {
    const int h = n0 >> 6;
#pragma unroll
    for (int r = 0; r < 8; ++r) {
      const int m = m0 + r + 8 * halfsel;
      const int b = m / Tc, t = m % Tc;
#pragma unroll
      for (int j = 0; j < 4; ++j)
        out16[(((size_t)b * Hc + h) * Dc + 16 * j + l15) * Tc + t] =
            (_Float16)acc[j][r];
    }
  } else {
#pragma unroll
    for (int r = 0; r < 8; ++r) {
      const int m = m0 + r + 8 * halfsel;
#pragma unroll
      for (int j = 0; j < 4; ++j)
        outf[(size_t)m * N + n0 + 16 * j + l15] = acc[j][r];
    }
  }
}

// ---------------------------------------------------------------------------
// Flash attention: one wave per (b, h, 16-query tile); stream 32-key chunks.
// qh,kh: (B,H,T,D) f16.  vT: (B,H,D,T) f16.  yh out: (B,T,C) f16.
// ---------------------------------------------------------------------------
__global__ void attn_kernel(const _Float16* __restrict__ qh,
                            const _Float16* __restrict__ kh,
                            const _Float16* __restrict__ vT,
                            _Float16* __restrict__ yh) {
  __shared__ _Float16 smem[4][16 * 32];   // per-wave P re-layout buffer
  const int lane    = threadIdx.x & 31;
  const int halfsel = lane >> 4;
  const int l15     = lane & 15;
  const int wv      = threadIdx.x >> 5;
  const int wid     = blockIdx.x * (blockDim.x >> 5) + wv;
  const int qt  = wid & 127;              // T/16 = 128 query tiles
  const int bh  = wid >> 7;               // 0..31 (b*H + h)
  const int tq0 = qt * 16;
  if (bh >= Bc * Hc) return;

  // Q A-fragments (16 x 64 = two K=32 slices), loaded once.
  const _Float16* qbase = qh + ((size_t)bh * Tc + tq0 + l15) * Dc;
  const v16h aq0 = load_frag(qbase, halfsel);
  const v16h aq1 = load_frag(qbase + 32, halfsel);

  v8f O[4];
  v8f zero = {};
#pragma unroll
  for (int j = 0; j < 4; ++j) O[j] = zero;
  float mrun[8], lrun[8];
#pragma unroll
  for (int r = 0; r < 8; ++r) { mrun[r] = -__builtin_inff(); lrun[r] = 0.f; }

  _Float16* sm = smem[wv];
  const int nck = (tq0 + 15) / 32 + 1;    // causal: keys up to tq0+15
  for (int ck = 0; ck < nck; ++ck) {
    const int kt0 = ck * 32;
    // S(16x32) = Q @ K^T : two 16-col score tiles, K-dim = D in two steps.
    v8f s0 = zero, s1 = zero;
    const _Float16* kb0 = kh + ((size_t)bh * Tc + kt0 + l15) * Dc;
    const _Float16* kb1 = kb0 + (size_t)16 * Dc;
    s0 = wmma_f16(aq0, load_frag(kb0, halfsel), s0);
    s0 = wmma_f16(aq1, load_frag(kb0 + 32, halfsel), s0);
    s1 = wmma_f16(aq0, load_frag(kb1, halfsel), s1);
    s1 = wmma_f16(aq1, load_frag(kb1 + 32, halfsel), s1);

    // Online softmax per query row (row lives across the 16-lane half).
#pragma unroll
    for (int r = 0; r < 8; ++r) {
      const int tq = tq0 + r + 8 * halfsel;
      float e0 = s0[r] * 0.125f;                 // 1/sqrt(64)
      if (kt0 + l15 > tq) e0 = -1e30f;
      float e1 = s1[r] * 0.125f;
      if (kt0 + 16 + l15 > tq) e1 = -1e30f;
      const float mx   = red_max16(fmaxf(e0, e1));
      const float mnew = fmaxf(mrun[r], mx);
      const float alpha = __expf(mrun[r] - mnew);
      const float p0 = __expf(e0 - mnew);
      const float p1 = __expf(e1 - mnew);
      const float ps = red_sum16(p0 + p1);
      lrun[r] = lrun[r] * alpha + ps;
      mrun[r] = mnew;
#pragma unroll
      for (int j = 0; j < 4; ++j) O[j][r] *= alpha;
      // C-layout -> LDS in plain row-major for A-fragment re-read.
      sm[(r + 8 * halfsel) * 32 + l15]      = (_Float16)p0;
      sm[(r + 8 * halfsel) * 32 + 16 + l15] = (_Float16)p1;
    }
    asm volatile("s_wait_dscnt 0" ::: "memory");
    const v16h pa = load_frag(sm + (size_t)l15 * 32, halfsel);

    // O(16x64) += P(16x32) @ V(32x64); vT rows are contiguous over t.
#pragma unroll
    for (int j = 0; j < 4; ++j) {
      const _Float16* vb = vT + ((size_t)bh * Dc + 16 * j + l15) * Tc + kt0;
      O[j] = wmma_f16(pa, load_frag(vb, halfsel), O[j]);
    }
  }

  const int b = bh >> 4, h = bh & 15;
#pragma unroll
  for (int r = 0; r < 8; ++r) {
    const int tq = tq0 + r + 8 * halfsel;
    const float inv = 1.0f / lrun[r];
#pragma unroll
    for (int j = 0; j < 4; ++j)
      yh[((size_t)b * Tc + tq) * Cc + h * Dc + 16 * j + l15] =
          (_Float16)(O[j][r] * inv);
  }
}

// ---------------------------------------------------------------------------
extern "C" void kernel_launch(void* const* d_in, const int* in_sizes, int n_in,
                              void* d_out, int out_size, void* d_ws,
                              size_t ws_size, hipStream_t stream) {
  const float* x    = (const float*)d_in[0];
  const float* cosp = (const float*)d_in[1];
  const float* sinp = (const float*)d_in[2];
  const float* Wq   = (const float*)d_in[3];
  const float* Wk   = (const float*)d_in[4];
  const float* Wv   = (const float*)d_in[5];
  const float* Wp   = (const float*)d_in[6];

  char* ws = (char*)d_ws;
  const size_t MB = (size_t)1 << 20;
  _Float16* xh  = (_Float16*)(ws + 0 * MB);    // 8 MB  (4M halfs)
  _Float16* wqT = (_Float16*)(ws + 8 * MB);    // 2 MB each
  _Float16* wkT = (_Float16*)(ws + 10 * MB);
  _Float16* wvT = (_Float16*)(ws + 12 * MB);
  _Float16* wpT = (_Float16*)(ws + 14 * MB);
  _Float16* qh  = (_Float16*)(ws + 16 * MB);   // 8 MB (B,H,T,D)
  _Float16* kh  = (_Float16*)(ws + 24 * MB);   // 8 MB (B,H,T,D)
  _Float16* vT  = (_Float16*)(ws + 32 * MB);   // 8 MB (B,H,D,T)
  _Float16* yh  = (_Float16*)(ws + 40 * MB);   // 8 MB (B,T,C)

  const int nx = Mc * Cc;
  cvt_f16_kernel<<<(nx + 255) / 256, 256, 0, stream>>>(x, xh, nx);
  const int nw = Cc * Cc;
  transpose_cvt_kernel<<<(nw + 255) / 256, 256, 0, stream>>>(Wq, wqT, Cc, Cc);
  transpose_cvt_kernel<<<(nw + 255) / 256, 256, 0, stream>>>(Wk, wkT, Cc, Cc);
  transpose_cvt_kernel<<<(nw + 255) / 256, 256, 0, stream>>>(Wv, wvT, Cc, Cc);
  transpose_cvt_kernel<<<(nw + 255) / 256, 256, 0, stream>>>(Wp, wpT, Cc, Cc);

  // (M/64)*(N/64) = 1024 blocks of 4 waves (64x64 block tile).
  const int gemmBlocks = (Mc / 64) * (Cc / 64);   // 1024
  gemm16_kernel<<<gemmBlocks, 128, 0, stream>>>(xh, wqT, qh, nullptr, cosp,
                                                sinp, Mc, Cc, Cc, /*mode=*/1);
  gemm16_kernel<<<gemmBlocks, 128, 0, stream>>>(xh, wkT, kh, nullptr, cosp,
                                                sinp, Mc, Cc, Cc, /*mode=*/1);
  gemm16_kernel<<<gemmBlocks, 128, 0, stream>>>(xh, wvT, vT, nullptr, nullptr,
                                                nullptr, Mc, Cc, Cc, /*mode=*/0);

  // B*H*(T/16) = 4096 wave tiles, 4 waves per block.
  const int attnBlocks = (Bc * Hc * (Tc / 16)) / 4;   // 1024
  attn_kernel<<<attnBlocks, 128, 0, stream>>>(qh, kh, vT, yh);

  gemm16_kernel<<<gemmBlocks, 128, 0, stream>>>(yh, wpT, nullptr, (float*)d_out,
                                                nullptr, nullptr, Mc, Cc, Cc,
                                                /*mode=*/2);
}
